// Seq2SeqRNN_11922829214046
// MI455X (gfx1250) — matrix operand order repaired and checked by reading, not compiled, not run on hardware
//
#include <hip/hip_runtime.h>
#include <hip/hip_bf16.h>

// Seq2Seq GRU (encoder + attention decoder) for MI455X / gfx1250.
// GEMMs: v_wmma_f32_16x16x32_bf16, A-tiles double-buffered in LDS via
// global_load_async_to_lds_b128 (ASYNCcnt) and read back with ds loads.
// Non-recurrent GEMMs batched across timesteps; only h@Whh.T and the
// ctx slice of the decoder input gates run per-step.

#define BB    32
#define SS    128
#define TTK   64
#define VV    32000
#define EE    512
#define HH    512
#define STEPS 63
#define GG    (3*HH)

typedef __bf16 bf16;
typedef bf16  v16bf __attribute__((ext_vector_type(16)));
typedef float v8f   __attribute__((ext_vector_type(8)));

__device__ __forceinline__ float sigf(float x) { return 1.0f / (1.0f + __expf(-x)); }

// B fragment: 32x16 (K x N) from weights stored [N, ldw] row-major.
// col n = lane%16; lanes 0-15 hold K=0..7,16..23, lanes 16-31 hold K=8..15,24..31 (ISA 7.12.2).
__device__ __forceinline__ v16bf frag_b(const bf16* W, int ldw, int n0, int kbase) {
  int l = threadIdx.x & 31;
  const bf16* r = W + (size_t)(n0 + (l & 15)) * ldw + kbase + 8 * (l >> 4);
  v16bf b;
#pragma unroll
  for (int e = 0; e < 16; ++e) b[e] = r[(e & 7) + ((e >> 3) << 4)];
  return b;
}

// ---------------- WMMA GEMM:  C[M,N] = A[M,K] * W[N, kofs:kofs+K]^T ----------------
// 256 threads = 8 waves; block tile 32(M) x 256(N); wave = 1 M-tile x 4 N-tiles.
// A k-tile (32x32 bf16, 2KB) is async-staged into double-buffered LDS by waves 0-3;
// every wave reuses each A fragment for 4 WMMAs.  W streams from global (L2).
// EPI 0: f32 -> Cf.  EPI 1: tanh(x+bias) -> bf16 Cb.  EPI 2: x+bias -> scatter logits.
template <int EPI>
__global__ __launch_bounds__(256) void gemm_k(
    const bf16* __restrict__ A, int lda,
    const bf16* __restrict__ W, int ldw, int kofs, int K,
    float* __restrict__ Cf, bf16* __restrict__ Cb, int ldc,
    const float* __restrict__ bias, float* __restrict__ scat)
{
  __shared__ bf16 Asm[2][32 * 32];
  int tid = threadIdx.x;
  int wv  = tid >> 5, l = tid & 31;
  int mt  = wv >> 2;                    // wave's M tile (0..1)
  int ntw = (wv & 3) * 64;              // wave's N base inside block
  int m0b = blockIdx.y * 32;
  int n0b = blockIdx.x * 256 + ntw;

  // Stage one 32x32 bf16 A k-tile into LDS buffer `buf` (128 x 16B async transfers).
  auto stage = [&](int buf, int k0) {
    if (tid < 128) {
      int r  = tid >> 2;                // row 0..31
      int cb = (tid & 3) * 16;          // byte column 0/16/32/48
      const char* g = (const char*)(A + (size_t)(m0b + r) * lda + k0) + cb;
      unsigned long long ga = (unsigned long long)(const void*)g;
      unsigned lo = (unsigned)(size_t)(&Asm[buf][0]) + (unsigned)(r * 64 + cb);
      asm volatile("global_load_async_to_lds_b128 %0, %1, off"
                   :: "v"(lo), "v"(ga) : "memory");
    }
  };

  stage(0, 0);
  v8f acc[4] = {};
  int p = 0;
  for (int k0 = 0; k0 < K; k0 += 32, p ^= 1) {
    asm volatile("s_wait_asynccnt 0" ::: "memory");   // issuing waves: LDS writes done
    __syncthreads();                                  // buf p visible; buf p^1 free
    if (k0 + 32 < K) stage(p ^ 1, k0 + 32);
    if (k0 + 32 < K)                                  // keep W stream warm in L2/L0
      __builtin_prefetch(W + (size_t)(n0b + (l & 15)) * ldw + kofs + k0 + 32, 0, 1);

    // A fragment from LDS (ISA 7.12.2 16-bit A layout).
    int hf = l >> 4;
    const bf16* ar = &Asm[p][(mt * 16 + (l & 15)) * 32];
    v16bf a;
#pragma unroll
    for (int e = 0; e < 8; ++e) a[e]     = ar[8 * hf + e];
#pragma unroll
    for (int e = 0; e < 8; ++e) a[8 + e] = ar[16 + 8 * hf + e];

#pragma unroll
    for (int q = 0; q < 4; ++q) {
      v16bf b = frag_b(W, ldw, n0b + q * 16, kofs + k0);
      acc[q] = __builtin_amdgcn_wmma_f32_16x16x32_bf16(false, a, false, b, (short)0,
                                                       acc[q], false, false);
    }
  }

  int mh = l >> 4;
#pragma unroll
  for (int q = 0; q < 4; ++q) {
    int n = n0b + q * 16 + (l & 15);
    float bs = (EPI == 0) ? 0.0f : bias[n];
#pragma unroll
    for (int v = 0; v < 8; ++v) {
      int m = m0b + mt * 16 + v + 8 * mh;   // D layout: VGPR v = row v + 8*(lane/16)
      float x = acc[q][v] + bs;
      if (EPI == 0) {
        Cf[(size_t)m * ldc + n] = x;
      } else if (EPI == 1) {
        Cb[(size_t)m * ldc + n] = (bf16)tanhf(x);
      } else {                               // logits scatter: row m = t*B + b
        int t = m >> 5, bi = m & 31;
        scat[((size_t)bi * STEPS + t) * VV + n] = x;
      }
    }
  }
}

// ---------------- small elementwise / setup kernels ----------------
__global__ void cvt_bf16_k(const float* __restrict__ s, bf16* __restrict__ d, int n) {
  int i = blockIdx.x * 256 + threadIdx.x;
  if (i < n) d[i] = (bf16)s[i];
}

__global__ void gather_enc_k(const int* __restrict__ src, const float* __restrict__ emb,
                             bf16* __restrict__ x) {
  int i = blockIdx.x * 256 + threadIdx.x;       // over S*B*E, exact
  int e = i % EE; int row = i / EE;             // row = s*B + b
  int b = row % BB, s = row / BB;
  int tok = src[b * SS + s];
  x[(size_t)row * EE + e] = (bf16)emb[(size_t)tok * EE + e];
}

__global__ void gather_dec_k(const int* __restrict__ tgt, const float* __restrict__ emb,
                             bf16* __restrict__ x) {
  int i = blockIdx.x * 256 + threadIdx.x;       // over STEPS*B*E, exact
  int e = i % EE; int row = i / EE;             // row = t*B + b
  int b = row % BB, t = row / BB;
  int tok = tgt[b * TTK + t];
  x[(size_t)row * EE + e] = (bf16)emb[(size_t)tok * EE + e];
}

__global__ void init_h_k(float* hf, bf16* hb) {
  int i = blockIdx.x * 256 + threadIdx.x;       // B*H, exact
  hf[i] = 0.0f; hb[i] = (bf16)0.0f;
}

__global__ void enc_update_k(const float* __restrict__ gi,   // pre-offset to step s
                             const float* __restrict__ gh,
                             const float* __restrict__ bih, const float* __restrict__ bhh,
                             const int* __restrict__ len,
                             float* __restrict__ h, bf16* __restrict__ hb,
                             float* __restrict__ enc_outs, int s)
{
  int i = blockIdx.x * 256 + threadIdx.x;       // B*H
  int j = i % HH, b = i / HH;
  const float* gir = gi + (size_t)b * GG;
  const float* ghr = gh + (size_t)b * GG;
  float ir = gir[j]          + bih[j];
  float iz = gir[HH + j]     + bih[HH + j];
  float in = gir[2 * HH + j] + bih[2 * HH + j];
  float hr = ghr[j]          + bhh[j];
  float hz = ghr[HH + j]     + bhh[HH + j];
  float hn = ghr[2 * HH + j] + bhh[2 * HH + j];
  float r  = sigf(ir + hr), z = sigf(iz + hz);
  float nn = tanhf(in + r * hn);
  float hold = h[i];
  float hnew = (1.0f - z) * nn + z * hold;
  bool valid = s < len[b];
  float hv = valid ? hnew : hold;
  h[i] = hv; hb[i] = (bf16)hv;
  enc_outs[((size_t)b * SS + s) * HH + j] = valid ? hnew : 0.0f;
}

__global__ void dec_update_k(const float* __restrict__ gi_emb,  // pre-offset to step t
                             const float* __restrict__ gi_ctx,
                             const float* __restrict__ gh,
                             const float* __restrict__ bih, const float* __restrict__ bhh,
                             float* __restrict__ h, bf16* __restrict__ hb,
                             bf16* __restrict__ cat, int t)
{
  int i = blockIdx.x * 256 + threadIdx.x;       // B*H
  int j = i % HH, b = i / HH;
  const float* gir = gi_emb + (size_t)b * GG;
  const float* gcr = gi_ctx + (size_t)b * GG;
  const float* ghr = gh + (size_t)b * GG;
  float ir = gir[j]          + gcr[j]          + bih[j];
  float iz = gir[HH + j]     + gcr[HH + j]     + bih[HH + j];
  float in = gir[2 * HH + j] + gcr[2 * HH + j] + bih[2 * HH + j];
  float hr = ghr[j]          + bhh[j];
  float hz = ghr[HH + j]     + bhh[HH + j];
  float hn = ghr[2 * HH + j] + bhh[2 * HH + j];
  float r  = sigf(ir + hr), z = sigf(iz + hz);
  float nn = tanhf(in + r * hn);
  float hnew = (1.0f - z) * nn + z * h[i];
  h[i] = hnew; hb[i] = (bf16)hnew;
  cat[((size_t)(t * BB + b)) * (2 * HH) + j] = (bf16)hnew;   // [hn, ctx] concat row
}

// Per-step attention: scores -> masked softmax -> context.  One block per batch b.
__global__ __launch_bounds__(256) void attn_k(
    const float* __restrict__ enc, const float* __restrict__ h,
    const int* __restrict__ len,
    float* __restrict__ ctxf, bf16* __restrict__ ctxb, bf16* __restrict__ cat,
    float* __restrict__ attn_out, int t)
{
  __shared__ float sc[SS];
  int b  = blockIdx.x;
  int wv = threadIdx.x >> 5, ln = threadIdx.x & 31;
  const float* hr = h + (size_t)b * HH;
  for (int s = wv; s < SS; s += 8) {            // 8 waves, 16 rows each
    const float* er = enc + ((size_t)b * SS + s) * HH;
    float p = 0.f;
    for (int k = ln; k < HH; k += 32) p += er[k] * hr[k];
#pragma unroll
    for (int off = 16; off > 0; off >>= 1) p += __shfl_xor(p, off, 32);
    if (ln == 0) sc[s] = p;
  }
  __syncthreads();
  if (wv == 0) {                                // wave 0: masked softmax over S=128
    int L = len[b];
    float v[4];
#pragma unroll
    for (int q = 0; q < 4; ++q) { int s = ln + 32 * q; v[q] = (s < L) ? sc[s] : -__builtin_inff(); }
    float m = fmaxf(fmaxf(v[0], v[1]), fmaxf(v[2], v[3]));
#pragma unroll
    for (int off = 16; off > 0; off >>= 1) m = fmaxf(m, __shfl_xor(m, off, 32));
    float e[4]; float sum = 0.f;
#pragma unroll
    for (int q = 0; q < 4; ++q) { e[q] = __expf(v[q] - m); sum += e[q]; }
#pragma unroll
    for (int off = 16; off > 0; off >>= 1) sum += __shfl_xor(sum, off, 32);
    float inv = 1.0f / sum;
#pragma unroll
    for (int q = 0; q < 4; ++q) {
      int s = ln + 32 * q;
      float a = e[q] * inv;
      sc[s] = a;
      attn_out[((size_t)b * STEPS + t) * SS + s] = a;
    }
  }
  __syncthreads();
  for (int j = threadIdx.x; j < HH; j += 256) { // ctx = aw @ enc_outs[b]
    float a = 0.f;
    for (int s = 0; s < SS; ++s) a += sc[s] * enc[((size_t)b * SS + s) * HH + j];
    ctxf[(size_t)b * HH + j] = a;
    ctxb[(size_t)b * HH + j] = (bf16)a;
    cat[((size_t)t * BB + b) * (2 * HH) + HH + j] = (bf16)a;
  }
}

extern "C" void kernel_launch(void* const* d_in, const int* in_sizes, int n_in,
                              void* d_out, int out_size, void* d_ws, size_t ws_size,
                              hipStream_t stream)
{
  (void)in_sizes; (void)n_in; (void)out_size; (void)ws_size;
  const int*   src   = (const int*)d_in[0];
  const int*   slen  = (const int*)d_in[1];
  const int*   tgt   = (const int*)d_in[2];
  const float* emb_e = (const float*)d_in[3];
  const float* Wih_e = (const float*)d_in[4];
  const float* Whh_e = (const float*)d_in[5];
  const float* bih_e = (const float*)d_in[6];
  const float* bhh_e = (const float*)d_in[7];
  const float* emb_d = (const float*)d_in[8];
  const float* Wih_d = (const float*)d_in[9];
  const float* Whh_d = (const float*)d_in[10];
  const float* bih_d = (const float*)d_in[11];
  const float* bhh_d = (const float*)d_in[12];
  const float* Wc    = (const float*)d_in[13];
  const float* bc    = (const float*)d_in[14];
  const float* Wo    = (const float*)d_in[15];
  const float* bo    = (const float*)d_in[16];

  // ---- workspace carve-up (~101 MB) ----
  char* ws = (char*)d_ws; size_t cur = 0;
  auto carve = [&](size_t bytes) -> char* {
    char* p = ws + cur; cur += (bytes + 255) & ~(size_t)255; return p;
  };
  bf16*  Wih_e_b = (bf16*)carve((size_t)GG * EE * 2);
  bf16*  Whh_e_b = (bf16*)carve((size_t)GG * HH * 2);
  bf16*  Wih_d_b = (bf16*)carve((size_t)GG * (EE + HH) * 2);
  bf16*  Whh_d_b = (bf16*)carve((size_t)GG * HH * 2);
  bf16*  Wc_b    = (bf16*)carve((size_t)HH * 2 * HH * 2);
  bf16*  Wo_b    = (bf16*)carve((size_t)VV * HH * 2);
  bf16*  x_enc_b = (bf16*)carve((size_t)SS * BB * EE * 2);
  bf16*  x_dec_b = (bf16*)carve((size_t)STEPS * BB * EE * 2);
  float* gi_enc  = (float*)carve((size_t)SS * BB * GG * 4);
  float* gi_dec  = (float*)carve((size_t)STEPS * BB * GG * 4);
  float* gh      = (float*)carve((size_t)BB * GG * 4);
  float* gi_ctx  = (float*)carve((size_t)BB * GG * 4);
  float* h_f     = (float*)carve((size_t)BB * HH * 4);
  bf16*  h_b     = (bf16*)carve((size_t)BB * HH * 2);
  float* enc_out = (float*)carve((size_t)BB * SS * HH * 4);
  float* ctx_f   = (float*)carve((size_t)BB * HH * 4);
  bf16*  ctx_b   = (bf16*)carve((size_t)BB * HH * 2);
  bf16*  cat_b   = (bf16*)carve((size_t)STEPS * BB * 2 * HH * 2);
  bf16*  cc_b    = (bf16*)carve((size_t)STEPS * BB * HH * 2);

  auto cvt = [&](const float* s, bf16* d, int n) {
    cvt_bf16_k<<<(n + 255) / 256, 256, 0, stream>>>(s, d, n);
  };
  cvt(Wih_e, Wih_e_b, GG * EE);
  cvt(Whh_e, Whh_e_b, GG * HH);
  cvt(Wih_d, Wih_d_b, GG * (EE + HH));
  cvt(Whh_d, Whh_d_b, GG * HH);
  cvt(Wc,    Wc_b,    HH * 2 * HH);
  cvt(Wo,    Wo_b,    VV * HH);

  gather_enc_k<<<(SS * BB * EE) / 256, 256, 0, stream>>>(src, emb_e, x_enc_b);
  gather_dec_k<<<(STEPS * BB * EE) / 256, 256, 0, stream>>>(tgt, emb_d, x_dec_b);
  init_h_k<<<(BB * HH) / 256, 256, 0, stream>>>(h_f, h_b);

  // Batched input-gate GEMMs (all timesteps at once); N block = 256
  gemm_k<0><<<dim3(GG / 256, (SS * BB) / 32), 256, 0, stream>>>(
      x_enc_b, EE, Wih_e_b, EE, 0, EE, gi_enc, nullptr, GG, nullptr, nullptr);
  gemm_k<0><<<dim3(GG / 256, (STEPS * BB) / 32), 256, 0, stream>>>(
      x_dec_b, EE, Wih_d_b, EE + HH, 0, EE, gi_dec, nullptr, GG, nullptr, nullptr);

  // ---- encoder recurrence ----
  for (int s = 0; s < SS; ++s) {
    gemm_k<0><<<dim3(GG / 256, 1), 256, 0, stream>>>(
        h_b, HH, Whh_e_b, HH, 0, HH, gh, nullptr, GG, nullptr, nullptr);
    enc_update_k<<<(BB * HH) / 256, 256, 0, stream>>>(
        gi_enc + (size_t)s * BB * GG, gh, bih_e, bhh_e, slen, h_f, h_b, enc_out, s);
  }

  // ---- decoder recurrence (h continues from encoder final state) ----
  float* attn_base = (float*)d_out + (size_t)BB * STEPS * VV;
  for (int t = 0; t < STEPS; ++t) {
    attn_k<<<BB, 256, 0, stream>>>(enc_out, h_f, slen, ctx_f, ctx_b, cat_b, attn_base, t);
    gemm_k<0><<<dim3(GG / 256, 1), 256, 0, stream>>>(
        h_b, HH, Whh_d_b, HH, 0, HH, gh, nullptr, GG, nullptr, nullptr);
    gemm_k<0><<<dim3(GG / 256, 1), 256, 0, stream>>>(
        ctx_b, HH, Wih_d_b, EE + HH, EE, HH, gi_ctx, nullptr, GG, nullptr, nullptr);
    dec_update_k<<<(BB * HH) / 256, 256, 0, stream>>>(
        gi_dec + (size_t)t * BB * GG, gi_ctx, gh, bih_d, bhh_d, h_f, h_b, cat_b, t);
  }

  // ---- deferred batched epilogue: cc = tanh([hn,ctx] @ Wc^T + bc), then logits ----
  gemm_k<1><<<dim3(HH / 256, (STEPS * BB) / 32), 256, 0, stream>>>(
      cat_b, 2 * HH, Wc_b, 2 * HH, 0, 2 * HH, nullptr, cc_b, HH, bc, nullptr);
  gemm_k<2><<<dim3(VV / 256, (STEPS * BB) / 32), 256, 0, stream>>>(
      cc_b, HH, Wo_b, HH, 0, HH, nullptr, nullptr, 0, bo, (float*)d_out);
}
